// MyLSTM_48009144435418
// MI455X (gfx1250) — matrix operand order repaired and checked by reading, compiled, not run on hardware
//
#include <hip/hip_runtime.h>

// ---------------------------------------------------------------------------
// LSTM on MI455X (gfx1250), bf16 WMMA (v_wmma_f32_16x16x32_bf16).
//   Phase 0: pack x, Wx, Wh (and h0) once into the exact CDNA5 per-lane
//            A-/B-fragment VGPR layouts (bf16) -> GEMM loops are pure b128
//            loads + WMMA (no LDS staging, no conversions).
//   Phase 1: xz = x @ Wx^T + b   (parallel, ~137 GFLOP)
//   Phase 2: 256 fused recurrent steps; software-pipelined fragment loads,
//            h' written back in packed fragment form for minimal step latency.
// ---------------------------------------------------------------------------

#define S_LEN 256
#define B_DIM 64
#define H_DIM 1024
#define G_DIM 4096 // 4*H

typedef __attribute__((ext_vector_type(16))) __bf16      v16bf;
typedef __attribute__((ext_vector_type(8)))  float        v8f;
typedef __attribute__((ext_vector_type(8)))  unsigned int v8u;

__device__ __forceinline__ unsigned f2bf(float f) {
  unsigned u = __builtin_bit_cast(unsigned, f);
  unsigned r = u + 0x7FFFu + ((u >> 16) & 1u); // RNE
  return r >> 16;
}
__device__ __forceinline__ float sigmoidf_(float x) {
  return 1.0f / (1.0f + __expf(-x));
}
__device__ __forceinline__ v8f wmma_bf16(v8u a, v8u b, v8f c) {
  return __builtin_amdgcn_wmma_f32_16x16x32_bf16(
      false, __builtin_bit_cast(v16bf, a),
      false, __builtin_bit_cast(v16bf, b),
      (short)0, c, false, false);
}

// ---------------------------------------------------------------------------
// Pack a row-major f32 activation matrix (rows x 1024) into bf16 16x32
// A-fragments. Fragment (mTile,kTile): lane L holds row m = mTile*16+(L&15);
// dwords 0-3 = K pairs 8*(L>>4)+{0,2,4,6}, dwords 4-7 = 16+8*(L>>4)+{0,2,4,6}.
// ---------------------------------------------------------------------------
__global__ void pack_a_kernel(const float* __restrict__ src, unsigned* __restrict__ dst,
                              int nMTiles) {
  const unsigned t = blockIdx.x * blockDim.x + threadIdx.x;
  const int lane = t & 31;
  const int fragIdx = t >> 5; // mTile*32 + kTile
  if (fragIdx >= nMTiles * 32) return;
  const int kTile = fragIdx & 31;
  const int mTile = fragIdx >> 5;
  const int hf = lane >> 4;
  const int m = mTile * 16 + (lane & 15);
  const float* row = src + (size_t)m * H_DIM + kTile * 32;
  unsigned* d = dst + (size_t)fragIdx * 256 + lane * 8;
#pragma unroll
  for (int v = 0; v < 8; ++v) {
    const int kit = (v < 4 ? 2 * v : 16 + 2 * (v - 4)) + 8 * hf;
    d[v] = f2bf(row[kit]) | (f2bf(row[kit + 1]) << 16);
  }
}

// ---------------------------------------------------------------------------
// Pack W (Wx or Wh half of [Wf;Wi;Wg;Wo]) into bf16 32x16 B-fragments.
// Fragment (nTile,kTile): lane L holds column n = nTile*16+(L&15);
// dwords 0-3 = K pairs kTile*32 + 8*(L>>4)+{0,2,4,6}, dwords 4-7 = kTile*32 + 16+8*(L>>4)+{0,2,4,6}.
// ---------------------------------------------------------------------------
__global__ void pack_w_kernel(const float* __restrict__ Wf, const float* __restrict__ Wi,
                              const float* __restrict__ Wg, const float* __restrict__ Wo,
                              unsigned* __restrict__ wxp, unsigned* __restrict__ whp) {
  const int isWh = blockIdx.y;
  const unsigned t = blockIdx.x * blockDim.x + threadIdx.x; // 0..262143
  const int lane = t & 31;
  const int fragIdx = t >> 5; // nTile*32 + kTile
  const int kTile = fragIdx & 31;
  const int nTile = fragIdx >> 5;
  const int khalf = lane >> 4;
  const int n_global = nTile * 16 + (lane & 15);
  const int q = n_global >> 10;
  const int r = n_global & 1023;
  const float* Wq = (q == 0) ? Wf : (q == 1) ? Wi : (q == 2) ? Wg : Wo;
  const float* row = Wq + (size_t)r * (2 * H_DIM) + (isWh ? H_DIM : 0);
  unsigned* dst = (isWh ? whp : wxp) + (size_t)fragIdx * 256 + lane * 8;
#pragma unroll
  for (int v = 0; v < 8; ++v) {
    const int kg = kTile * 32 + (v < 4 ? 2 * v : 16 + 2 * (v - 4)) + 8 * khalf;
    dst[v] = f2bf(row[kg]) | (f2bf(row[kg + 1]) << 16);
  }
}

__global__ void pack_bias_kernel(const float* __restrict__ bf, const float* __restrict__ bi,
                                 const float* __restrict__ bg, const float* __restrict__ bo,
                                 float* __restrict__ ball) {
  const int t = blockIdx.x * blockDim.x + threadIdx.x;
  if (t < G_DIM) {
    const int q = t >> 10, r = t & 1023;
    ball[t] = (q == 0) ? bf[r] : (q == 1) ? bi[r] : (q == 2) ? bg[r] : bo[r];
  }
}

// ---------------------------------------------------------------------------
// Phase 1 GEMM: xz[row,g] = sum_k x[row,k]*Wx[g,k] + b[g]
// M=16384, N=4096, K=1024. Software-pipelined fragment loads (double-
// buffered registers) so WMMAs overlap the next iteration's b128 loads.
// ---------------------------------------------------------------------------
__global__ void __launch_bounds__(128)
gemm_xz_kernel(const unsigned* __restrict__ xp, const unsigned* __restrict__ wxp,
               const float* __restrict__ ball, float* __restrict__ xz) {
  const int tid = threadIdx.x;
  const int lane = tid & 31;
  const int w = tid >> 5;
  const int bx = blockIdx.x; // 64-row block
  const int by = blockIdx.y; // 64-col block
  const int khalf = lane >> 4;
  const v8u* ap = (const v8u*)xp;
  const v8u* bp = (const v8u*)wxp;
  const size_t abase = (size_t)(bx * 4 + w) * 32 * 32 + lane;
  const size_t bbase = (size_t)(by * 4) * 32 * 32 + lane;

  v8f acc[4] = {};
  // prologue: fragments for kt = 0
  v8u a_cur = ap[abase];
  v8u b_cur[4];
#pragma unroll
  for (int f = 0; f < 4; ++f) b_cur[f] = bp[bbase + (size_t)(f * 32) * 32];

#pragma unroll 2
  for (int kt = 0; kt < 32; ++kt) {
    const int kn = (kt + 1) & 31; // wrap: final iteration's loads are dead
    v8u a_nxt = ap[abase + (size_t)kn * 32];
    v8u b_nxt[4];
#pragma unroll
    for (int f = 0; f < 4; ++f) b_nxt[f] = bp[bbase + (size_t)(f * 32 + kn) * 32];
#pragma unroll
    for (int f = 0; f < 4; ++f) acc[f] = wmma_bf16(a_cur, b_cur[f], acc[f]);
    a_cur = a_nxt;
#pragma unroll
    for (int f = 0; f < 4; ++f) b_cur[f] = b_nxt[f];
  }

  // C layout: lane holds column n=(L&15), rows v + 8*(L>>4)
#pragma unroll
  for (int f = 0; f < 4; ++f) {
    const int n = by * 64 + f * 16 + (lane & 15);
    const float bias = ball[n];
#pragma unroll
    for (int v = 0; v < 8; ++v) {
      const size_t row = (size_t)bx * 64 + w * 16 + v + 8 * khalf;
      xz[row * G_DIM + n] = acc[f][v] + bias;
    }
  }
}

// ---------------------------------------------------------------------------
// One fused recurrent step:
//   z = xz[s] + h @ Wh^T ; gates ; c' = f*c + i*g ; h' = o*tanh(c')
//   out[s] = x[s] + h' ; h' re-packed to bf16 A-fragments for next step.
// Grid (4 batch-tiles, 64 hidden-tiles), 128 threads = 4 waves (1 per gate).
// Two interleaved WMMA chains + double-buffered fragment registers.
// ---------------------------------------------------------------------------
__global__ void __launch_bounds__(128)
lstm_step_kernel(const float* __restrict__ x, const float* __restrict__ xz,
                 const unsigned* __restrict__ whp,
                 const unsigned* __restrict__ hp_in, unsigned* __restrict__ hp_out,
                 float* __restrict__ c_buf, float* __restrict__ out,
                 float* __restrict__ hfin, float* __restrict__ cfin,
                 int s, int is_last) {
  __shared__ float zbuf[4][16][16]; // 4 KB gate-tile exchange
  const int tid = threadIdx.x;
  const int lane = tid & 31;
  const int w = tid >> 5;    // gate
  const int mt = blockIdx.x; // batch tile
  const int jt = blockIdx.y; // hidden-column tile
  const int m0 = mt * 16;
  const int khalf = lane >> 4;
  const v8u* ap = (const v8u*)hp_in;
  const v8u* bp = (const v8u*)whp;
  const int nTile = w * 64 + jt;
  const size_t abase = (size_t)mt * 32 * 32 + lane;
  const size_t bbase = (size_t)nTile * 32 * 32 + lane;

  v8f acc0 = {}, acc1 = {};
  // prologue: kt = 0,1
  v8u a0c = ap[abase];
  v8u b0c = bp[bbase];
  v8u a1c = ap[abase + 32];
  v8u b1c = bp[bbase + 32];
#pragma unroll 2
  for (int kt = 0; kt < 32; kt += 2) {
    const int kn = (kt + 2) & 31; // wrap: final loads are dead
    const v8u a0n = ap[abase + (size_t)kn * 32];
    const v8u b0n = bp[bbase + (size_t)kn * 32];
    const v8u a1n = ap[abase + (size_t)(kn + 1) * 32];
    const v8u b1n = bp[bbase + (size_t)(kn + 1) * 32];
    acc0 = wmma_bf16(a0c, b0c, acc0);
    acc1 = wmma_bf16(a1c, b1c, acc1);
    a0c = a0n; b0c = b0n; a1c = a1n; b1c = b1n;
  }

  // z = acc + xz, exchange gate tiles via LDS
  const int n_local = lane & 15;
  const int ng = w * H_DIM + jt * 16 + n_local;
#pragma unroll
  for (int v = 0; v < 8; ++v) {
    const int ml = v + 8 * khalf;
    const size_t row = (size_t)s * B_DIM + m0 + ml;
    zbuf[w][ml][n_local] = acc0[v] + acc1[v] + xz[row * G_DIM + ng];
  }
  __syncthreads();

  // elementwise gates: 256 (m,j) cells, 128 threads x 2
#pragma unroll
  for (int i = 0; i < 2; ++i) {
    const int e = tid + i * 128;
    const int m = e >> 4, j = e & 15;
    const float zf = zbuf[0][m][j];
    const float zi = zbuf[1][m][j];
    const float zg = zbuf[2][m][j];
    const float zo = zbuf[3][m][j];
    const int b = m0 + m;
    const int jg = jt * 16 + j;
    const size_t cidx = (size_t)b * H_DIM + jg;
    const float fg = sigmoidf_(zf);
    const float ig = sigmoidf_(zi);
    const float gg = tanhf(zg);
    const float og = sigmoidf_(zo);
    const float cc = fg * c_buf[cidx] + ig * gg;
    const float hh = og * tanhf(cc);
    c_buf[cidx] = cc;
    const size_t orow = ((size_t)s * B_DIM + b) * H_DIM + jg;
    out[orow] = x[orow] + hh;
    if (is_last) { hfin[cidx] = hh; cfin[cidx] = cc; }

    // scatter h' into packed bf16 A-fragment layout for the next step:
    // element (row m, K=jg) of fragment (mTile=mt, kTile=jg>>5)
    const int kin = jg & 31;
    const int klow = kin & 15;
    const int hfl = klow >> 3; // K-half -> lane group
    const int vv = (kin >= 16 ? 4 : 0) + ((klow - 8 * hfl) >> 1);
    const int lane16 = m + 16 * hfl;
    const size_t hidx = ((size_t)(mt * 32 + (jg >> 5)) * 512) + lane16 * 16 + vv * 2 + (kin & 1);
    ((unsigned short*)hp_out)[hidx] = (unsigned short)f2bf(hh);
  }
}

// ---------------------------------------------------------------------------
// Workspace layout (bytes):
//   xz  f32 S*B*4H                 : 256 MB
//   WxP bf16 B-frags               :   8 MB
//   WhP bf16 B-frags               :   8 MB
//   xP  bf16 A-frags (1024x32)     :  32 MB
//   b_all f32                      :  16 KB
//   hP ping-pong bf16 A-frags (2x) : 256 KB
//   c f32 B*H                      : 256 KB
// ---------------------------------------------------------------------------
static const size_t XZ_OFF  = 0;
static const size_t WXP_OFF = XZ_OFF  + (size_t)S_LEN * B_DIM * G_DIM * 4;
static const size_t WHP_OFF = WXP_OFF + (size_t)256 * 32 * 1024;
static const size_t XP_OFF  = WHP_OFF + (size_t)256 * 32 * 1024;
static const size_t BAL_OFF = XP_OFF  + (size_t)1024 * 32 * 1024;
static const size_t HP0_OFF = BAL_OFF + (size_t)G_DIM * 4;
static const size_t HP1_OFF = HP0_OFF + (size_t)4 * 32 * 1024;
static const size_t C_OFF   = HP1_OFF + (size_t)4 * 32 * 1024;

extern "C" void kernel_launch(void* const* d_in, const int* in_sizes, int n_in,
                              void* d_out, int out_size, void* d_ws, size_t ws_size,
                              hipStream_t stream) {
  const float* x  = (const float*)d_in[0];
  const float* h0 = (const float*)d_in[1];
  const float* c0 = (const float*)d_in[2];
  const float* Wf = (const float*)d_in[3];
  const float* bf = (const float*)d_in[4];
  const float* Wi = (const float*)d_in[5];
  const float* bi = (const float*)d_in[6];
  const float* Wg = (const float*)d_in[7];
  const float* bg = (const float*)d_in[8];
  const float* Wo = (const float*)d_in[9];
  const float* bo = (const float*)d_in[10];

  char* ws = (char*)d_ws;
  float*    xz   = (float*)(ws + XZ_OFF);
  unsigned* wxp  = (unsigned*)(ws + WXP_OFF);
  unsigned* whp  = (unsigned*)(ws + WHP_OFF);
  unsigned* xp   = (unsigned*)(ws + XP_OFF);
  float*    ball = (float*)(ws + BAL_OFF);
  unsigned* hp0  = (unsigned*)(ws + HP0_OFF);
  unsigned* hp1  = (unsigned*)(ws + HP1_OFF);
  float*    cb   = (float*)(ws + C_OFF);

  float* out  = (float*)d_out;
  float* hfin = out + (size_t)S_LEN * B_DIM * H_DIM;
  float* cfin = hfin + (size_t)B_DIM * H_DIM;

  // pack weights / bias / x / h0 into WMMA fragment layouts
  pack_w_kernel<<<dim3(1024, 2), 256, 0, stream>>>(Wf, Wi, Wg, Wo, wxp, whp);
  pack_bias_kernel<<<16, 256, 0, stream>>>(bf, bi, bg, bo, ball);
  pack_a_kernel<<<4096, 256, 0, stream>>>(x, xp, 1024); // 16384 rows
  pack_a_kernel<<<16, 256, 0, stream>>>(h0, hp0, 4);    // 64 rows
  hipMemcpyAsync(cb, c0, (size_t)B_DIM * H_DIM * 4, hipMemcpyDeviceToDevice, stream);

  // parallel precompute: xz = x @ Wx^T + b
  gemm_xz_kernel<<<dim3(256, 64), 128, 0, stream>>>(xp, wxp, ball, xz);

  // serial scan: one fused kernel per timestep, packed h double-buffered
  for (int s = 0; s < S_LEN; ++s) {
    const unsigned* hin = (s & 1) ? hp1 : hp0;
    unsigned*       hou = (s & 1) ? hp0 : hp1;
    lstm_step_kernel<<<dim3(4, 64), 128, 0, stream>>>(
        x, xz, whp, hin, hou, cb, out, hfin, cfin, s, (s == S_LEN - 1) ? 1 : 0);
  }
}
